// OuterProductLayer_2602750182019
// MI455X (gfx1250) — matrix operand order repaired and checked by reading, not compiled, not run on hardware
//
#include <hip/hip_runtime.h>

#define EMB 64
#define NF 32
#define NPAIR 496
#define NPAIR_PAD 512
#define D1 256
#define BATCH 2048

typedef __attribute__((ext_vector_type(16))) __bf16        v16bf;
typedef __attribute__((ext_vector_type(8)))  float         v8f;
typedef __attribute__((ext_vector_type(4)))  float         v4f;
typedef __attribute__((ext_vector_type(2)))  unsigned int  v2u;
typedef __attribute__((ext_vector_type(4)))  unsigned int  v4u;
typedef __attribute__((ext_vector_type(8)))  unsigned int  v8u;

__device__ __forceinline__ unsigned int f2bf_bits(float f) {
  unsigned int u = __builtin_bit_cast(unsigned int, f);
  u += 0x7FFFu + ((u >> 16) & 1u);          // round-to-nearest-even
  return u >> 16;
}
__device__ __forceinline__ float bf2f(unsigned short h) {
  return __builtin_bit_cast(float, (unsigned int)h << 16);
}

// ---------------- conversion / init kernels ----------------

// 4 elements per thread: float4 load -> packed 2x u32 (4 bf16) store
__global__ void cvt_bf16_v4_kernel(const float* __restrict__ in,
                                   unsigned short* __restrict__ out, int n4) {
  int i = blockIdx.x * blockDim.x + threadIdx.x;
  if (i >= n4) return;
  v4f x = *(const v4f*)(in + (size_t)i * 4);
  v2u r;
  r[0] = f2bf_bits(x[0]) | (f2bf_bits(x[1]) << 16);
  r[1] = f2bf_bits(x[2]) | (f2bf_bits(x[3]) << 16);
  *(v2u*)(out + (size_t)i * 4) = r;
}

// W [D1][NPAIR] f32 -> bf16 [D1][NPAIR_PAD] zero-padded
__global__ void cvt_w_pad_kernel(const float* __restrict__ W,
                                 unsigned short* __restrict__ out) {
  int i = blockIdx.x * blockDim.x + threadIdx.x;   // over D1*NPAIR_PAD
  int n = i >> 9, k = i & (NPAIR_PAD - 1);
  out[i] = (k < NPAIR) ? (unsigned short)f2bf_bits(W[n * NPAIR + k])
                       : (unsigned short)0;
}

// zero the pad columns [NPAIR..NPAIR_PAD) of pij
__global__ void pad_pij_kernel(unsigned short* __restrict__ pij) {
  int i = blockIdx.x * blockDim.x + threadIdx.x;   // over BATCH*16
  int b = i >> 4, k = NPAIR + (i & 15);
  pij[(size_t)b * NPAIR_PAD + k] = 0;
}

// ---------------- kernel 2: per-pair bilinear form via WMMA ----------------
// One wave: one pair x 8 batch tiles (128 rows).  The pair's 64x64 bf16 kernel
// slice lives in 8 resident B fragments (64 VGPRs).  The tile loop is
// software-pipelined: tile t's 8 WMMAs issue while tile t-1's accumulators are
// consumed by the fused q-dot + 16-lane reduction, hiding the WMMA->VALU
// hazard slots with independent matrix work.
__global__ void __launch_bounds__(256)
bilinear_pair_kernel(const unsigned short* __restrict__ featb, // bf16 [B][NF][EMB]
                     const unsigned short* __restrict__ kernb, // bf16 [d][NPAIR][e]
                     unsigned short*       __restrict__ pij)   // bf16 [B][NPAIR_PAD]
{
  const int p = blockIdx.x;
  // pair -> (rowp, colp); uniform across block -> SALU loop
  int fi = 0, rem = p;
  while (rem >= (NF - 1 - fi)) { rem -= (NF - 1 - fi); ++fi; }
  const int rowp = fi, colp = fi + 1 + rem;

  const int wave = threadIdx.x >> 5;
  const int lane = threadIdx.x & 31;
  const int lm   = lane & 15;
  const int lh   = lane >> 4;
  const int slot = blockIdx.y * 8 + wave;          // 0..15; 8 tiles each

  // ---- preload the 8 resident B fragments (d-block j, K-step ks) ----
  v16bf bfr[2][4];
#pragma unroll
  for (int ks = 0; ks < 2; ++ks) {
#pragma unroll
    for (int j = 0; j < 4; ++j) {
      const int d  = 16 * j + lm;                  // column for this lane
      const int e0 = 32 * ks + 8 * lh;             // K = e0..e0+7, e0+16..e0+23
      const unsigned short* bp = kernb + ((size_t)d * NPAIR + p) * EMB + e0;
      v4u b_lo = *(const v4u*)(bp);
      v4u b_hi = *(const v4u*)(bp + 16);
      v8u bu;
      bu[0] = b_lo[0]; bu[1] = b_lo[1]; bu[2] = b_lo[2]; bu[3] = b_lo[3];
      bu[4] = b_hi[0]; bu[5] = b_hi[1]; bu[6] = b_hi[2]; bu[7] = b_hi[3];
      bfr[ks][j] = __builtin_bit_cast(v16bf, bu);
    }
  }

  // T-tile for batch tile t: A(16x64 e) x B -> acc[4] (16x64 d)
  auto compute_tile = [&](int t, v8f acc[4]) {
    const int b0 = (slot * 8 + t) * 16;
    const unsigned short* arow = featb + ((size_t)(b0 + lm) * NF + rowp) * EMB;
#pragma unroll
    for (int j = 0; j < 4; ++j) acc[j] = v8f{};
#pragma unroll
    for (int ks = 0; ks < 2; ++ks) {
      const int kb = 32 * ks + 8 * lh;
      v4u a_lo = *(const v4u*)(arow + kb);         // K = kb..kb+7
      v4u a_hi = *(const v4u*)(arow + kb + 16);    // K = kb+16..kb+23
      v8u au;
      au[0] = a_lo[0]; au[1] = a_lo[1]; au[2] = a_lo[2]; au[3] = a_lo[3];
      au[4] = a_hi[0]; au[5] = a_hi[1]; au[6] = a_hi[2]; au[7] = a_hi[3];
      v16bf afrag = __builtin_bit_cast(v16bf, au);
#pragma unroll
      for (int j = 0; j < 4; ++j) {
        acc[j] = __builtin_amdgcn_wmma_f32_16x16x32_bf16(
                     false, afrag, false, bfr[ks][j], (short)0, acc[j],
                     false, false);
      }
    }
  };

  // pij[b] = sum_d T[b,d]*q[b,d]; acc reg r, lane -> row r+8*lh, d = 16j+lm
  auto reduce_tile = [&](int t, v8f acc[4]) {
    const int b0 = (slot * 8 + t) * 16;
#pragma unroll
    for (int r = 0; r < 8; ++r) {
      const int b = b0 + r + 8 * lh;
      const unsigned short* qrow = featb + ((size_t)b * NF + colp) * EMB;
      float s = acc[0][r] * bf2f(qrow[lm])
              + acc[1][r] * bf2f(qrow[16 + lm])
              + acc[2][r] * bf2f(qrow[32 + lm])
              + acc[3][r] * bf2f(qrow[48 + lm]);
      s += __shfl_xor(s, 1);
      s += __shfl_xor(s, 2);
      s += __shfl_xor(s, 4);
      s += __shfl_xor(s, 8);                       // reduce within 16-lane half
      if (lm == 0) pij[(size_t)b * NPAIR_PAD + p] = (unsigned short)f2bf_bits(s);
    }
  };

  v8f accA[4], accB[4];
  compute_tile(0, accA);
#pragma unroll
  for (int t = 1; t < 8; ++t) {
    v8f* cur  = (t & 1) ? accB : accA;
    v8f* prev = (t & 1) ? accA : accB;
    compute_tile(t, cur);                          // independent WMMA work...
    reduce_tile(t - 1, prev);                      // ...covers this consumer
  }
  reduce_tile(7, accB);
}

// ---------------- kernel 3: out = pij @ W^T + b via WMMA ----------------
__global__ void __launch_bounds__(256)
final_gemm_kernel(const unsigned short* __restrict__ pijb, // bf16 [B][512]
                  const unsigned short* __restrict__ wb,   // bf16 [D1][512]
                  const float*          __restrict__ bias, // f32  [D1]
                  float*                __restrict__ out)  // f32  [B][D1]
{
  const int wave = threadIdx.x >> 5;
  const int lane = threadIdx.x & 31;
  const int tid  = blockIdx.x * 8 + wave;          // 0..2047 output tiles
  const int mt   = tid >> 4;                       // 0..127 (batch tile)
  const int nt   = tid & 15;                       // 0..15  (D1 tile)
  const int lm   = lane & 15;
  const int lh   = lane >> 4;

  const unsigned short* arow = pijb + (size_t)(mt * 16 + lm) * NPAIR_PAD;
  const unsigned short* brow = wb   + (size_t)(nt * 16 + lm) * NPAIR_PAD;

  v8f acc = {};
#pragma unroll
  for (int kk = 0; kk < 16; ++kk) {                // K = 512 in steps of 32
    const int ka = kk * 32 + 8 * lh;
    v4u a_lo = *(const v4u*)(arow + ka);
    v4u a_hi = *(const v4u*)(arow + ka + 16);
    v8u au;
    au[0] = a_lo[0]; au[1] = a_lo[1]; au[2] = a_lo[2]; au[3] = a_lo[3];
    au[4] = a_hi[0]; au[5] = a_hi[1]; au[6] = a_hi[2]; au[7] = a_hi[3];
    v16bf afrag = __builtin_bit_cast(v16bf, au);
    v4u b_lo = *(const v4u*)(brow + ka);
    v4u b_hi = *(const v4u*)(brow + ka + 16);
    v8u bu;
    bu[0] = b_lo[0]; bu[1] = b_lo[1]; bu[2] = b_lo[2]; bu[3] = b_lo[3];
    bu[4] = b_hi[0]; bu[5] = b_hi[1]; bu[6] = b_hi[2]; bu[7] = b_hi[3];
    v16bf bfrag = __builtin_bit_cast(v16bf, bu);
    acc = __builtin_amdgcn_wmma_f32_16x16x32_bf16(
              false, afrag, false, bfrag, (short)0, acc, false, false);
  }

  const float bn = bias[nt * 16 + lm];
#pragma unroll
  for (int r = 0; r < 8; ++r) {
    const int m = mt * 16 + r + 8 * lh;
    out[(size_t)m * D1 + nt * 16 + lm] = acc[r] + bn;
  }
}

// ---------------- host launcher ----------------

extern "C" void kernel_launch(void* const* d_in, const int* in_sizes, int n_in,
                              void* d_out, int out_size, void* d_ws, size_t ws_size,
                              hipStream_t stream) {
  (void)in_sizes; (void)n_in; (void)out_size; (void)ws_size;

  const float* feat = (const float*)d_in[0];   // [2048][32][64]
  const float* kern = (const float*)d_in[1];   // [64][496][64]
  const float* W    = (const float*)d_in[2];   // [256][496]
  const float* bias = (const float*)d_in[3];   // [256]
  float*       out  = (float*)d_out;           // [2048][256]

  char* ws = (char*)d_ws;
  const size_t FEAT_B = (size_t)BATCH * NF * EMB * 2;      // 8,388,608
  const size_t KERN_B = (size_t)EMB * NPAIR * EMB * 2;     // 4,063,232
  const size_t W_B    = (size_t)D1 * NPAIR_PAD * 2;        //   262,144
  unsigned short* featb = (unsigned short*)(ws);
  unsigned short* kernb = (unsigned short*)(ws + FEAT_B);
  unsigned short* wb    = (unsigned short*)(ws + FEAT_B + KERN_B);
  unsigned short* pijb  = (unsigned short*)(ws + FEAT_B + KERN_B + W_B);

  const int nfeat4 = (BATCH * NF * EMB) / 4;   // 1,048,576
  const int nkern4 = (EMB * NPAIR * EMB) / 4;  //   507,904
  cvt_bf16_v4_kernel<<<nfeat4 / 256, 256, 0, stream>>>(feat, featb, nfeat4);
  cvt_bf16_v4_kernel<<<nkern4 / 256, 256, 0, stream>>>(kern, kernb, nkern4);
  cvt_w_pad_kernel<<<(D1 * NPAIR_PAD) / 256, 256, 0, stream>>>(W, wb);
  pad_pij_kernel<<<(BATCH * 16) / 256, 256, 0, stream>>>(pijb);

  dim3 g2(NPAIR, 2);                // 496 pairs x (16 wave-slots / 8 waves)
  bilinear_pair_kernel<<<g2, 256, 0, stream>>>(featb, kernb, pijb);

  final_gemm_kernel<<<BATCH / 8, 256, 0, stream>>>(pijb, wb, bias, out);
}
